// DualStreamNodeDetector_13400297964060
// MI455X (gfx1250) — matrix operand order repaired and checked by reading, not compiled, not run on hardware
//
#include <hip/hip_runtime.h>
#include <hip/hip_bf16.h>
#include <stdint.h>

typedef __attribute__((ext_vector_type(16))) __bf16 v16bf;
typedef __attribute__((ext_vector_type(8)))  float  v8f;

union BFrag {
  uint4 q[2];
  v16bf v;
  unsigned short s[16];
};

__device__ __forceinline__ unsigned short f2bf(float f) {
  unsigned int u = __float_as_uint(f);
  u += 0x7FFFu + ((u >> 16) & 1u);      // round-to-nearest-even
  return (unsigned short)(u >> 16);
}
__device__ __forceinline__ unsigned int pk2(float a, float b) {
  return (unsigned int)f2bf(a) | ((unsigned int)f2bf(b) << 16);
}

__device__ __forceinline__ unsigned ldsoff(const void* p) {
  // __shared__ flat address = shared-aperture-base (4GB aligned) | lds offset
  return (unsigned)(unsigned long long)p;
}
__device__ __forceinline__ void async_b128(unsigned lds, const void* g) {
  asm volatile("global_load_async_to_lds_b128 %0, %1, off"
               :: "v"(lds), "v"(g) : "memory");
}
__device__ __forceinline__ void wait_async4() {
  asm volatile("s_wait_asynccnt 0x4" ::: "memory");
}
__device__ __forceinline__ void wait_async0() {
  asm volatile("s_wait_asynccnt 0x0" ::: "memory");
}

#define LDSROW 40   // 32 bf16 data + 8 pad -> 80B row stride, conflict-free b128 reads

// ---------------------------------------------------------------------------
// GEMM: C[M,Ncols] = A[M,512](bf16) @ W (+bias); W pre-transposed bf16
// Bt[Ncols,512].  Block = 256 thr (8 waves), tile = 128 rows x 128 cols.
// A/B K-chunks (32) staged via async global->LDS, double-buffered.
// ---------------------------------------------------------------------------
__global__ __launch_bounds__(256)
void k_gemm_bf16(const unsigned short* __restrict__ A,
                 const unsigned short* __restrict__ Bt,
                 const float* __restrict__ bias, float* __restrict__ C,
                 int M, int Ncols) {
  __shared__ __align__(16) unsigned short lA[2][128 * LDSROW];
  __shared__ __align__(16) unsigned short lB[2][128 * LDSROW];

  const int tid   = threadIdx.x;
  const int wave  = tid >> 5;
  const int lane  = tid & 31;
  const int mlane = lane & 15;
  const int hsel  = lane >> 4;
  const int tb_row = blockIdx.y * 128;
  const int tb_col = blockIdx.x * 128;

  v8f acc[8] = {v8f{}, v8f{}, v8f{}, v8f{}, v8f{}, v8f{}, v8f{}, v8f{}};

  // staging: 2 threads per row, each copies 32B (2 x b128) of the 64B row chunk
  const int srow = tid >> 1;
  const int sseg = (tid & 1) << 5;                 // byte offset 0 or 32
  const int grow = tb_row + srow;
  const int growc = (grow < M) ? grow : (M - 1);   // clamp: uniform async count
  const char* gA = (const char*)(A + (size_t)growc * 512) + sseg;
  const char* gB = (const char*)(Bt + (size_t)(tb_col + srow) * 512) + sseg;
  unsigned aoff[2], boff[2];
  aoff[0] = ldsoff(&lA[0][srow * LDSROW]) + sseg;
  aoff[1] = ldsoff(&lA[1][srow * LDSROW]) + sseg;
  boff[0] = ldsoff(&lB[0][srow * LDSROW]) + sseg;
  boff[1] = ldsoff(&lB[1][srow * LDSROW]) + sseg;

  auto issue = [&](int k0, int b) {
    const char* ga = gA + k0 * 2;
    const char* gb = gB + k0 * 2;
    async_b128(aoff[b],      ga);
    async_b128(aoff[b] + 16, ga + 16);
    async_b128(boff[b],      gb);
    async_b128(boff[b] + 16, gb + 16);
  };

  issue(0, 0);
  int pb = 0;
  for (int k0 = 0; k0 < 512; k0 += 32, pb ^= 1) {
    if (k0 + 32 < 512) { issue(k0 + 32, pb ^ 1); wait_async4(); }
    else               { wait_async0(); }
    __syncthreads();

    // A fragment (ISA 16-bit A 16x32): elems 0..7 -> K=hsel*8+j, 8..15 -> 16+hsel*8+j
    const unsigned short* ap = &lA[pb][(wave * 16 + mlane) * LDSROW];
    BFrag af;
    af.q[0] = *(const uint4*)(ap + hsel * 8);
    af.q[1] = *(const uint4*)(ap + 16 + hsel * 8);

#pragma unroll
    for (int ct = 0; ct < 8; ++ct) {
      // B fragment: N = mlane, elems 0..7 -> K=hsel*8+j, 8..15 -> 16+hsel*8+j
      const unsigned short* bp = &lB[pb][(ct * 16 + mlane) * LDSROW + hsel * 8];
      BFrag bfr;
      bfr.q[0] = *(const uint4*)(bp);
      bfr.q[1] = *(const uint4*)(bp + 16);
      acc[ct] = __builtin_amdgcn_wmma_f32_16x16x32_bf16(
          false, af.v, false, bfr.v, (short)0, acc[ct], false, false);
    }
    __syncthreads();   // all waves done reading before this buffer is re-filled
  }

  // Epilogue: C/D layout -> M = e + 8*hsel, N = mlane
#pragma unroll
  for (int ct = 0; ct < 8; ++ct) {
    const int col = tb_col + ct * 16 + mlane;
    const float bv = bias ? bias[col] : 0.0f;
#pragma unroll
    for (int e = 0; e < 8; ++e) {
      const int row = tb_row + wave * 16 + hsel * 8 + e;
      if (row < M) C[(size_t)row * Ncols + col] = acc[ct][e] + bv;
    }
  }
}

// ---------------------------------------------------------------------------
// Support kernels
// ---------------------------------------------------------------------------
__global__ void k_cvtT(const float* __restrict__ W, unsigned short* __restrict__ Wt,
                       int Ncols, int total) {   // W[512][Ncols] -> Wt[Ncols][512] bf16
  int id = blockIdx.x * blockDim.x + threadIdx.x;
  if (id >= total) return;
  int k = id / Ncols, n = id % Ncols;
  Wt[(size_t)n * 512 + k] = f2bf(W[id]);
}

// row-major f32 -> bf16 (activations), 4 elems/thread
__global__ void k_cvt(const float* __restrict__ in, unsigned short* __restrict__ out, int n4) {
  int id = blockIdx.x * blockDim.x + threadIdx.x;
  if (id >= n4) return;
  float4 f = ((const float4*)in)[id];
  uint2 o;
  o.x = pk2(f.x, f.y);
  o.y = pk2(f.z, f.w);
  ((uint2*)out)[id] = o;
}

__global__ void k_fill(float* __restrict__ p, float v, int n) {
  int i = blockIdx.x * blockDim.x + threadIdx.x;
  if (i < n) p[i] = v;
}

__global__ void k_deg(const int* __restrict__ dst, float* __restrict__ deg, int E) {
  int i = blockIdx.x * blockDim.x + threadIdx.x;
  if (i < E) atomicAdd(&deg[dst[i]], 1.0f);
}

__global__ void k_rsqrt(float* __restrict__ p, int n) {
  int i = blockIdx.x * blockDim.x + threadIdx.x;
  if (i < n) p[i] = rsqrtf(p[i]);
}

// agg = P * dinv^2 + bias   (self-loop + GCN bias)
__global__ void k_self(const float* __restrict__ P, const float* __restrict__ dinv,
                       const float* __restrict__ bias, float* __restrict__ agg, int n) {
  int id = blockIdx.x * blockDim.x + threadIdx.x;
  if (id >= n) return;
  int node = id >> 9;
  int f = id & 511;
  float di = dinv[node];
  agg[id] = P[id] * di * di + bias[f];
}

// agg[dst] += P[src] * dinv[src]*dinv[dst]
__global__ void k_edges(const float* __restrict__ P, const float* __restrict__ dinv,
                        const int* __restrict__ src, const int* __restrict__ dst,
                        float* __restrict__ agg, int total) {   // total = E*128
  int id = blockIdx.x * blockDim.x + threadIdx.x;
  if (id >= total) return;
  int e = id >> 7;
  int q = id & 127;
  int s = src[e], d = dst[e];
  float c = dinv[s] * dinv[d];
  float4 p = ((const float4*)(P + (size_t)s * 512))[q];
  float* o = agg + (size_t)d * 512 + (q << 2);
  atomicAdd(o + 0, p.x * c);
  atomicAdd(o + 1, p.y * c);
  atomicAdd(o + 2, p.z * c);
  atomicAdd(o + 3, p.w * c);
}

__global__ void k_relu(float* __restrict__ p, int n) {
  int i = blockIdx.x * blockDim.x + threadIdx.x;
  if (i < n) p[i] = fmaxf(p[i], 0.0f);
}

__global__ __launch_bounds__(256)
void k_bnstats(const float* __restrict__ T, float* __restrict__ stats, int N) {
  int c0 = threadIdx.x, c1 = threadIdx.x + 256;
  float s0 = 0.f, s1 = 0.f, q0 = 0.f, q1 = 0.f;
  for (int r = blockIdx.x; r < N; r += gridDim.x) {
    float a = T[(size_t)r * 512 + c0];
    float b = T[(size_t)r * 512 + c1];
    s0 += a; q0 += a * a;
    s1 += b; q1 += b * b;
  }
  atomicAdd(&stats[c0], s0);
  atomicAdd(&stats[c1], s1);
  atomicAdd(&stats[512 + c0], q0);
  atomicAdd(&stats[512 + c1], q1);
}

__global__ void k_bnfin(float* __restrict__ stats, const float* __restrict__ gamma,
                        const float* __restrict__ beta, float invN) {
  int f = blockIdx.x * blockDim.x + threadIdx.x;
  if (f >= 512) return;
  float mu  = stats[f] * invN;
  float var = stats[512 + f] * invN - mu * mu;
  float sc  = rsqrtf(var + 1e-5f) * gamma[f];
  stats[1024 + f] = sc;
  stats[1536 + f] = beta[f] - mu * sc;
}

__global__ void k_bnapply(float* __restrict__ T, const float* __restrict__ stats, int n) {
  int id = blockIdx.x * blockDim.x + threadIdx.x;
  if (id >= n) return;
  int f = id & 511;
  float y = T[id] * stats[1024 + f] + stats[1536 + f];
  T[id] = fmaxf(y, 0.0f);
}

__global__ __launch_bounds__(256)
void k_l2norm(const float* __restrict__ Zin, float* __restrict__ out, int N) {
  int wave = threadIdx.x >> 5, lane = threadIdx.x & 31;
  int row = blockIdx.x * 8 + wave;
  if (row >= N) return;
  float4 p = ((const float4*)(Zin + (size_t)row * 128))[lane];
  float ss = p.x * p.x + p.y * p.y + p.z * p.z + p.w * p.w;
  for (int o = 16; o > 0; o >>= 1) ss += __shfl_xor(ss, o, 32);
  float s = 1.0f / fmaxf(sqrtf(ss), 1e-12f);
  float4 r = make_float4(p.x * s, p.y * s, p.z * s, p.w * s);
  ((float4*)(out + (size_t)row * 128))[lane] = r;
}

// ---------------------------------------------------------------------------
extern "C" void kernel_launch(void* const* d_in, const int* in_sizes, int n_in,
                              void* d_out, int out_size, void* d_ws, size_t ws_size,
                              hipStream_t stream) {
  const float* x_topo = (const float*)d_in[0];
  const float* h_sem  = (const float*)d_in[1];
  const int*   edge   = (const int*)d_in[2];
  const float* W1  = (const float*)d_in[3];
  const float* b1  = (const float*)d_in[4];
  const float* W2  = (const float*)d_in[5];
  const float* b2  = (const float*)d_in[6];
  const float* Wt1 = (const float*)d_in[7];
  const float* bt1 = (const float*)d_in[8];
  const float* gt  = (const float*)d_in[9];
  const float* bt  = (const float*)d_in[10];
  const float* Wt2 = (const float*)d_in[11];
  const float* bt2 = (const float*)d_in[12];
  const float* Ws1 = (const float*)d_in[13];
  const float* bs1 = (const float*)d_in[14];
  const float* gs  = (const float*)d_in[15];
  const float* bs  = (const float*)d_in[16];
  const float* Ws2 = (const float*)d_in[17];
  const float* bs2 = (const float*)d_in[18];

  const int H = 512, ZD = 128;
  const int N = in_sizes[0] / H;
  const int E = in_sizes[2] / 2;
  const int* src = edge;
  const int* dst = edge + E;

  char* ws = (char*)d_ws;
  const size_t NB = (size_t)N * H * sizeof(float);
  float* buf0 = (float*)(ws);
  float* buf1 = (float*)(ws + NB);
  float* buf2 = (float*)(ws + 2 * NB);
  unsigned short* w1t  = (unsigned short*)(ws + 3 * NB);
  unsigned short* w2t  = w1t  + 512 * 512;
  unsigned short* wt1t = w2t  + 512 * 512;
  unsigned short* ws1t = wt1t + 512 * 512;
  unsigned short* wt2t = ws1t + 512 * 512;
  unsigned short* ws2t = wt2t + 128 * 512;
  float* dinv  = (float*)(ws2t + 128 * 512);
  float* stats = dinv + N;
  unsigned short* abf = (unsigned short*)(stats + 2048);   // bf16 activations [N,512]

  const dim3 b256(256);
  auto gfor = [](long long n) { return dim3((unsigned)((n + 255) / 256)); };

  const int NH  = N * H;        // 25.6M
  const int NH4 = NH / 4;
  const int ET  = E * 128;

  // 1) weights -> bf16, transposed [Ncols][K]
  k_cvtT<<<gfor(512 * 512), b256, 0, stream>>>(W1,  w1t,  512, 512 * 512);
  k_cvtT<<<gfor(512 * 512), b256, 0, stream>>>(W2,  w2t,  512, 512 * 512);
  k_cvtT<<<gfor(512 * 512), b256, 0, stream>>>(Wt1, wt1t, 512, 512 * 512);
  k_cvtT<<<gfor(512 * 512), b256, 0, stream>>>(Ws1, ws1t, 512, 512 * 512);
  k_cvtT<<<gfor(512 * 128), b256, 0, stream>>>(Wt2, wt2t, 128, 512 * 128);
  k_cvtT<<<gfor(512 * 128), b256, 0, stream>>>(Ws2, ws2t, 128, 512 * 128);

  // 2) dinv = rsqrt(deg + 1)
  k_fill<<<gfor(N), b256, 0, stream>>>(dinv, 1.0f, N);
  k_deg<<<gfor(E), b256, 0, stream>>>(dst, dinv, E);
  k_rsqrt<<<gfor(N), b256, 0, stream>>>(dinv, N);

  const dim3 g512(512 / 128, (N + 127) / 128);
  const dim3 g128(1, (N + 127) / 128);
  const dim3 gL2((N + 7) / 8);

  // 3) GCN layer 1
  k_cvt<<<gfor(NH4), b256, 0, stream>>>(x_topo, abf, NH4);
  k_gemm_bf16<<<g512, b256, 0, stream>>>(abf, w1t, nullptr, buf0, N, 512);
  k_self<<<gfor(NH), b256, 0, stream>>>(buf0, dinv, b1, buf1, NH);
  k_edges<<<gfor(ET), b256, 0, stream>>>(buf0, dinv, src, dst, buf1, ET);
  k_relu<<<gfor(NH), b256, 0, stream>>>(buf1, NH);

  // 4) GCN layer 2
  k_cvt<<<gfor(NH4), b256, 0, stream>>>(buf1, abf, NH4);
  k_gemm_bf16<<<g512, b256, 0, stream>>>(abf, w2t, nullptr, buf0, N, 512);
  k_self<<<gfor(NH), b256, 0, stream>>>(buf0, dinv, b2, buf2, NH);
  k_edges<<<gfor(ET), b256, 0, stream>>>(buf0, dinv, src, dst, buf2, ET);

  // 5) topo head
  k_cvt<<<gfor(NH4), b256, 0, stream>>>(buf2, abf, NH4);
  k_gemm_bf16<<<g512, b256, 0, stream>>>(abf, wt1t, bt1, buf1, N, 512);
  k_fill<<<gfor(2048), b256, 0, stream>>>(stats, 0.0f, 2048);
  k_bnstats<<<dim3(512), b256, 0, stream>>>(buf1, stats, N);
  k_bnfin<<<dim3(2), b256, 0, stream>>>(stats, gt, bt, 1.0f / (float)N);
  k_bnapply<<<gfor(NH), b256, 0, stream>>>(buf1, stats, NH);
  k_cvt<<<gfor(NH4), b256, 0, stream>>>(buf1, abf, NH4);
  k_gemm_bf16<<<g128, b256, 0, stream>>>(abf, wt2t, bt2, buf0, N, 128);
  k_l2norm<<<gL2, b256, 0, stream>>>(buf0, (float*)d_out, N);

  // 6) sem head
  k_cvt<<<gfor(NH4), b256, 0, stream>>>(h_sem, abf, NH4);
  k_gemm_bf16<<<g512, b256, 0, stream>>>(abf, ws1t, bs1, buf2, N, 512);
  k_fill<<<gfor(2048), b256, 0, stream>>>(stats, 0.0f, 2048);
  k_bnstats<<<dim3(512), b256, 0, stream>>>(buf2, stats, N);
  k_bnfin<<<dim3(2), b256, 0, stream>>>(stats, gs, bs, 1.0f / (float)N);
  k_bnapply<<<gfor(NH), b256, 0, stream>>>(buf2, stats, NH);
  k_cvt<<<gfor(NH4), b256, 0, stream>>>(buf2, abf, NH4);
  k_gemm_bf16<<<g128, b256, 0, stream>>>(abf, ws2t, bs2, buf1, N, 128);
  k_l2norm<<<gL2, b256, 0, stream>>>(buf1, (float*)d_out + (size_t)N * ZD, N);
}
